// MiMoV2FlashTopKRouter_36679020708355
// MI455X (gfx1250) — matrix operand order repaired and checked by reading, not compiled, not run on hardware
//
#include <hip/hip_runtime.h>
#include <math.h>

#define T_TOK 16384
#define HID   4096
#define NEXP  256
#define TOPK  8
#define NGRP  8
#define TGRP  4

typedef __attribute__((ext_vector_type(16))) __bf16 v16bf;
typedef __attribute__((ext_vector_type(8)))  __bf16 v8bf;
typedef __attribute__((ext_vector_type(4)))  __bf16 v4bf;
typedef __attribute__((ext_vector_type(8)))  float  v8f;
typedef __attribute__((ext_vector_type(4)))  float  f32x4;

// ---------------------------------------------------------------------------
// Kernel 0: one-time convert router weight [E,H] f32 -> bf16 (row-major) in ws.
// 1M elements, 4 per thread.
// ---------------------------------------------------------------------------
__global__ __launch_bounds__(256)
void cvt_weight_bf16(const float* __restrict__ W, __bf16* __restrict__ Wb) {
  size_t i = ((size_t)blockIdx.x * blockDim.x + threadIdx.x) * 4;
  f32x4 f = *(const f32x4*)(W + i);
  v4bf b;
#pragma unroll
  for (int j = 0; j < 4; ++j) b[j] = (__bf16)f[j];
  *(v4bf*)(Wb + i) = b;
}

// ---------------------------------------------------------------------------
// Kernel 1: logits = X[T,H] @ Wb[E,H]^T  via v_wmma_f32_16x16x32_bf16
// Block = 256 threads (8 waves). Block handles 64 tokens x 256 experts.
// Wave w: token tile (w & 3) (16 tokens), expert half (w >> 2) (8 tiles of 16).
// Per k-chunk: load+cvt A frag (f32->bf16), load 8 bf16 B frags, then 8
// back-to-back WMMAs on distinct accumulators (no inter-WMMA hazards).
// ---------------------------------------------------------------------------
__global__ __launch_bounds__(256)
void router_gemm_bf16(const float* __restrict__ X,
                      const __bf16* __restrict__ Wb,
                      float* __restrict__ L) {
  const int lane = threadIdx.x & 31;
  const int wave = threadIdx.x >> 5;
  const int tt   = wave & 3;          // token tile within block
  const int eh   = wave >> 2;         // expert half (0/1)
  const int tokBase = blockIdx.x * 64 + tt * 16;
  const int expBase = eh * 128;
  const int hi  = lane >> 4;          // half-wave id (0: lanes 0-15, 1: 16-31)
  const int lid = lane & 15;

  const float* xrow = X + (size_t)(tokBase + lid) * HID;

  v8f acc[8];
#pragma unroll
  for (int i = 0; i < 8; ++i) acc[i] = (v8f)0.0f;

#pragma unroll 2
  for (int k0 = 0; k0 < HID; k0 += 32) {
    // ---- A fragment (fp32 -> bf16 on the fly), ISA 16-bit A 16x32 layout:
    //  lanes 0-15 : K = k0..k0+7 (elems 0..7), k0+16..k0+23 (elems 8..15)
    //  lanes 16-31: K = k0+8..k0+15,           k0+24..k0+31
    const int kA = k0 + hi * 8;
    f32x4 a0 = *(const f32x4*)(xrow + kA);
    f32x4 a1 = *(const f32x4*)(xrow + kA + 4);
    f32x4 a2 = *(const f32x4*)(xrow + kA + 16);
    f32x4 a3 = *(const f32x4*)(xrow + kA + 20);
    v16bf afrag;
#pragma unroll
    for (int i = 0; i < 4; ++i) {
      afrag[i]      = (__bf16)a0[i];
      afrag[4 + i]  = (__bf16)a1[i];
      afrag[8 + i]  = (__bf16)a2[i];
      afrag[12 + i] = (__bf16)a3[i];
    }

    // ---- B fragments: lane = expert column; same K order as A
    //  lanes 0-15: K = k0..k0+7, k0+16..k0+23 ; lanes 16-31: K = k0+8..k0+15, k0+24..k0+31
    const int kB = k0 + hi * 8;
    v16bf bfrag[8];
#pragma unroll
    for (int et = 0; et < 8; ++et) {
      const __bf16* wrow = Wb + (size_t)(expBase + et * 16 + lid) * HID + kB;
      v8bf b0 = *(const v8bf*)(wrow);
      v8bf b1 = *(const v8bf*)(wrow + 16);
      bfrag[et] = __builtin_shufflevector(b0, b1,
          0, 1, 2, 3, 4, 5, 6, 7, 8, 9, 10, 11, 12, 13, 14, 15);
    }

    // ---- 8 WMMAs, distinct accumulators -> no data hazards between them
#pragma unroll
    for (int et = 0; et < 8; ++et) {
      acc[et] = __builtin_amdgcn_wmma_f32_16x16x32_bf16(
          false, afrag, false, bfrag[et], (short)0, acc[et], false, false);
    }
  }

  // ---- store: C/D layout — lane gives N = lid, VGPR r gives M = r + 8*hi
  const int outTok = tokBase + hi * 8;
#pragma unroll
  for (int et = 0; et < 8; ++et) {
    const int e = expBase + et * 16 + lid;
#pragma unroll
    for (int r = 0; r < 8; ++r) {
      L[(size_t)(outTok + r) * NEXP + e] = acc[et][r];
    }
  }
}

// ---------------------------------------------------------------------------
// Kernel 2: grouped sigmoid top-k routing. One wave32 per token.
// Each lane owns 8 experts (lane*8 .. lane*8+7); 4 lanes form one group of 32.
// ---------------------------------------------------------------------------
__global__ __launch_bounds__(256)
void router_topk(const float* __restrict__ L,
                 const float* __restrict__ bias,
                 float* __restrict__ topw,
                 int* __restrict__ topi) {
  const int lane = threadIdx.x & 31;
  const int wave = threadIdx.x >> 5;
  const int t = blockIdx.x * 8 + wave;
  if (t >= T_TOK) return;

  const float NEG = -__builtin_inff();
  const float* lrow = L + (size_t)t * NEXP;

  float sc[8], s4c[8];
#pragma unroll
  for (int i = 0; i < 8; ++i) {
    float x = lrow[lane * 8 + i];
    float s = 1.0f / (1.0f + __expf(-x));
    sc[i]  = s;
    s4c[i] = s + bias[lane * 8 + i];
  }

  // per-lane top-2 of biased scores
  float m1 = NEG, m2 = NEG;
#pragma unroll
  for (int i = 0; i < 8; ++i) {
    float v = s4c[i];
    if (v > m1) { m2 = m1; m1 = v; }
    else if (v > m2) { m2 = v; }
  }
  // merge top-2 across the 4 lanes of this group (butterfly)
#pragma unroll
  for (int off = 1; off < 4; off <<= 1) {
    float o1 = __shfl_xor(m1, off, 32);
    float o2 = __shfl_xor(m2, off, 32);
    if (o1 > m1) { m2 = fmaxf(m1, o2); m1 = o1; }
    else         { m2 = fmaxf(m2, o1); }
  }
  float gscore = m1 + m2;   // replicated across the 4 lanes of each group

  // gather the 8 group scores (group g's score lives in lane 4g)
  float gs[8];
#pragma unroll
  for (int g = 0; g < 8; ++g) gs[g] = __shfl(gscore, g * 4, 32);

  // select top-4 groups (every lane computes the same mask)
  unsigned gmask = 0;
#pragma unroll
  for (int r = 0; r < TGRP; ++r) {
    int best = 0; float bv = NEG;
#pragma unroll
    for (int g = 0; g < 8; ++g) {
      bool taken = (gmask >> g) & 1u;
      if (!taken && gs[g] > bv) { bv = gs[g]; best = g; }
    }
    gmask |= (1u << best);
  }

  const bool keep = (gmask >> (lane >> 2)) & 1u;
  float cand[8];
#pragma unroll
  for (int i = 0; i < 8; ++i) cand[i] = keep ? s4c[i] : NEG;

  // extract top-8 experts by repeated wave-wide argmax
  float wv[8]; int wi[8]; float wsum = 0.0f;
#pragma unroll
  for (int r = 0; r < TOPK; ++r) {
    float bv = NEG; int bi = 0;
#pragma unroll
    for (int i = 0; i < 8; ++i) {
      if (cand[i] > bv) { bv = cand[i]; bi = i; }
    }
    int   gidx = lane * 8 + bi;
    float bsc  = (bi == 0 ? sc[0] : bi == 1 ? sc[1] : bi == 2 ? sc[2] :
                  bi == 3 ? sc[3] : bi == 4 ? sc[4] : bi == 5 ? sc[5] :
                  bi == 6 ? sc[6] : sc[7]);
#pragma unroll
    for (int off = 16; off > 0; off >>= 1) {
      float ov = __shfl_xor(bv,   off, 32);
      int   oi = __shfl_xor(gidx, off, 32);
      float os = __shfl_xor(bsc,  off, 32);
      if (ov > bv || (ov == bv && oi < gidx)) { bv = ov; gidx = oi; bsc = os; }
    }
    wv[r] = bsc; wi[r] = gidx; wsum += bsc;
    // winning lane retires its candidate (static-index form, no scratch)
    const bool mine = (gidx >> 3) == lane;
    const int  bsl  = gidx & 7;
#pragma unroll
    for (int i = 0; i < 8; ++i) {
      if (mine && bsl == i) cand[i] = NEG;
    }
  }

  const float inv = 1.0f / (wsum + 1e-20f);  // ROUTED_SCALING_FACTOR == 1.0
  if (lane == 0) {
#pragma unroll
    for (int r = 0; r < TOPK; ++r) {
      topw[(size_t)t * TOPK + r] = wv[r] * inv;
      topi[(size_t)t * TOPK + r] = wi[r];
    }
  }
}

// ---------------------------------------------------------------------------
extern "C" void kernel_launch(void* const* d_in, const int* in_sizes, int n_in,
                              void* d_out, int out_size, void* d_ws, size_t ws_size,
                              hipStream_t stream) {
  (void)in_sizes; (void)n_in; (void)out_size; (void)ws_size;
  const float* X    = (const float*)d_in[0];   // [16384, 4096] f32
  const float* W    = (const float*)d_in[1];   // [256, 4096]   f32
  const float* bias = (const float*)d_in[2];   // [256]         f32

  float* out    = (float*)d_out;
  float* logits = out;                                   // T*E f32
  float* topw   = out + (size_t)T_TOK * NEXP;            // T*K f32
  int*   topi   = (int*)(topw + (size_t)T_TOK * TOPK);   // T*K i32 (bit-cast)

  __bf16* Wb = (__bf16*)d_ws;                            // 256*4096 bf16 = 2 MB

  cvt_weight_bf16<<<(NEXP * HID) / (256 * 4), 256, 0, stream>>>(W, Wb);
  router_gemm_bf16<<<T_TOK / 64, 256, 0, stream>>>(X, Wb, logits);
  router_topk<<<T_TOK / 8, 256, 0, stream>>>(logits, bias, topw, topi);
}
